// AttentionBlock_76252849373512
// MI455X (gfx1250) — matrix operand order repaired and checked by reading, not compiled, not run on hardware
//
#include <hip/hip_runtime.h>
#include <hip/hip_bf16.h>

typedef __attribute__((ext_vector_type(16))) __bf16 v16bf;
typedef __attribute__((ext_vector_type(8)))  __bf16 v8bf;
typedef __attribute__((ext_vector_type(8)))  float  v8f;
typedef __attribute__((ext_vector_type(4)))  float  v4f;

#define B_   16
#define C_   512
#define N_   1024
#define G_   32
#define CG_  16          // C_/G_
#define O3_  1536        // 3*C_

// ---------- WMMA helpers (contiguous k order, same for A and B, wave32) ----------
// A (16-bit, 16x32): lane half h holds K[16h..16h+15] contiguous, row M = lane&15.
static __device__ __forceinline__ v16bf load_a16(const __bf16* row, int k0, int half) {
  return *(const v16bf*)(row + k0 + half * 16);
}
// B (16-bit, 32x16): lane half h holds K[16h..16h+15] contiguous, col N = lane&15.
static __device__ __forceinline__ v16bf load_b16(const __bf16* row, int k0, int half) {
  return *(const v16bf*)(row + k0 + half * 16);
}
static __device__ __forceinline__ v8f wmma_bf16(v16bf a, v16bf b, v8f c) {
  return __builtin_amdgcn_wmma_f32_16x16x32_bf16(false, a, false, b, (short)0, c, false, false);
}

// ---------- Kernel 0: fp32 weights -> bf16 ----------
__global__ void k_cvt_weights(const float* __restrict__ wq, const float* __restrict__ wo,
                              __bf16* __restrict__ wqb, __bf16* __restrict__ wob) {
  const int nq = O3_ * C_;
  const int tot = nq + C_ * C_;
  for (int i = blockIdx.x * blockDim.x + threadIdx.x; i < tot; i += gridDim.x * blockDim.x) {
    if (i < nq) wqb[i] = (__bf16)wq[i];
    else        wob[i - nq] = (__bf16)wo[i - nq];
  }
}

// ---------- Kernel 1: GroupNorm stats (each (b,g) is a contiguous 16K-float slab) ----------
__global__ void k_gn_stats(const float* __restrict__ x, float* __restrict__ meanp,
                           float* __restrict__ rstdp) {
  const int bg = blockIdx.x;                       // b*G_+g
  const float* base = x + (size_t)bg * (CG_ * N_);
  float s = 0.f, s2 = 0.f;
  for (int i = threadIdx.x; i < CG_ * N_; i += blockDim.x) {
    float v = base[i]; s += v; s2 += v * v;
  }
  __shared__ float a1[256], a2[256];
  a1[threadIdx.x] = s; a2[threadIdx.x] = s2;
  __syncthreads();
  for (int st = 128; st > 0; st >>= 1) {
    if (threadIdx.x < st) { a1[threadIdx.x] += a1[threadIdx.x + st]; a2[threadIdx.x] += a2[threadIdx.x + st]; }
    __syncthreads();
  }
  if (threadIdx.x == 0) {
    const float inv = 1.0f / (CG_ * N_);
    float m = a1[0] * inv;
    float var = a2[0] * inv - m * m;
    meanp[bg] = m;
    rstdp[bg] = rsqrtf(var + 1e-5f);
  }
}

// ---------- Kernel 2: normalize + transpose -> xnT[b][n][c] (bf16) ----------
__global__ void k_norm_transpose(const float* __restrict__ x, const float* __restrict__ meanp,
                                 const float* __restrict__ rstdp, const float* __restrict__ gscale,
                                 const float* __restrict__ gbias, __bf16* __restrict__ xnT) {
  const int t  = blockIdx.x;
  const int nt = t & 31;            // token tile   (N_/32)
  const int ct = (t >> 5) & 15;     // channel tile (C_/32)
  const int b  = t >> 9;
  const int n0 = nt * 32, c0 = ct * 32;
  __shared__ __bf16 lds[32][33];
  const int col = threadIdx.x & 31;
  const int r0  = threadIdx.x >> 5;
#pragma unroll
  for (int i = 0; i < 4; ++i) {
    const int row = r0 + 8 * i;                   // channel-local
    const int ch  = c0 + row;
    const int g   = ch >> 4;
    const float v = x[((size_t)b * C_ + ch) * N_ + n0 + col];
    const float m = meanp[b * G_ + g], rs = rstdp[b * G_ + g];
    lds[row][col] = (__bf16)(((v - m) * rs) * gscale[ch] + gbias[ch]);
  }
  __syncthreads();
#pragma unroll
  for (int i = 0; i < 4; ++i) {
    const int tk = r0 + 8 * i;                    // token-local
    xnT[((size_t)b * N_ + n0 + tk) * C_ + c0 + col] = lds[col][tk];
  }
}

// ---------- Kernel 3: QKV GEMM (2x2 register-blocked) ----------
// qkvT[b][t][o] = W[o][:].xnT[b][t][:] + bias ; each wave does 32(o) x 32(t).
__global__ void k_qkv_gemm(const __bf16* __restrict__ wqb, const __bf16* __restrict__ xnT,
                           const float* __restrict__ bqkv, __bf16* __restrict__ qkvT) {
  const int wid  = blockIdx.x * (blockDim.x >> 5) + (threadIdx.x >> 5);
  const int lane = threadIdx.x & 31, half = lane >> 4, l16 = lane & 15;
  const int op = wid % 48;                  // 1536/32
  const int tp = (wid / 48) & 31;           // 1024/32
  const int b  = wid / (48 * 32);
  const int o0 = op * 32, t0 = tp * 32;
  const __bf16* ar0 = wqb + (size_t)(o0 + l16) * C_;             // A: M = o
  const __bf16* ar1 = ar0 + (size_t)16 * C_;
  const __bf16* br0 = xnT + ((size_t)b * N_ + t0 + l16) * C_;    // B: N = token
  const __bf16* br1 = br0 + (size_t)16 * C_;
  v8f c00 = {}, c01 = {}, c10 = {}, c11 = {};
#pragma unroll
  for (int k = 0; k < C_; k += 32) {
    v16bf a0 = load_a16(ar0, k, half);
    v16bf a1 = load_a16(ar1, k, half);
    v16bf b0 = load_b16(br0, k, half);
    v16bf b1 = load_b16(br1, k, half);
    c00 = wmma_bf16(a0, b0, c00);
    c01 = wmma_bf16(a0, b1, c01);
    c10 = wmma_bf16(a1, b0, c10);
    c11 = wmma_bf16(a1, b1, c11);
  }
  const float* bp0 = bqkv + o0 + half * 8;
  const float* bp1 = bp0 + 16;
  v8bf v00, v01, v10, v11;
#pragma unroll
  for (int r = 0; r < 8; ++r) {
    v00[r] = (__bf16)(c00[r] + bp0[r]);
    v01[r] = (__bf16)(c01[r] + bp0[r]);
    v10[r] = (__bf16)(c10[r] + bp1[r]);
    v11[r] = (__bf16)(c11[r] + bp1[r]);
  }
  __bf16* d0 = qkvT + ((size_t)b * N_ + t0 + l16) * O3_ + o0 + half * 8;
  __bf16* d1 = qkvT + ((size_t)b * N_ + t0 + 16 + l16) * O3_ + o0 + half * 8;
  *(v8bf*)(d0)      = v00;
  *(v8bf*)(d1)      = v01;
  *(v8bf*)(d0 + 16) = v10;
  *(v8bf*)(d1 + 16) = v11;
}

// ---------- Kernel 4: transpose V slice -> VT[b][c][n] ----------
__global__ void k_vt(const __bf16* __restrict__ qkvT, __bf16* __restrict__ VT) {
  const int t  = blockIdx.x;
  const int nt = t & 31;
  const int ct = (t >> 5) & 15;
  const int b  = t >> 9;
  const int n0 = nt * 32, c0 = ct * 32;
  __shared__ __bf16 lds[32][33];
  const int col = threadIdx.x & 31;
  const int r0  = threadIdx.x >> 5;
#pragma unroll
  for (int i = 0; i < 4; ++i) {
    const int row = r0 + 8 * i;                   // token-local
    lds[row][col] = qkvT[((size_t)b * N_ + n0 + row) * O3_ + 2 * C_ + c0 + col];
  }
  __syncthreads();
#pragma unroll
  for (int i = 0; i < 4; ++i) {
    const int ch = r0 + 8 * i;                    // channel-local
    VT[((size_t)b * C_ + c0 + ch) * N_ + n0 + col] = lds[col][ch];
  }
}

// ---------- Kernel 5: scores + stable softmax -> P (bf16, unnormalized), rowsum ----------
// Q A-tiles (16 x v16bf = 128 VGPRs) are hoisted into registers: one 32B B-fetch per WMMA.
__global__ void k_attn_softmax(const __bf16* __restrict__ qkvT, __bf16* __restrict__ P,
                               float* __restrict__ rowsum) {
  const int wave = threadIdx.x >> 5;
  const int wid  = blockIdx.x * 4 + wave;
  const int lane = threadIdx.x & 31, half = lane >> 4, l16 = lane & 15;
  const int qt = wid & 63, b = wid >> 6;
  const int q0 = qt * 16;
  const __bf16* base = qkvT + (size_t)b * N_ * O3_;
  const __bf16* arow = base + (size_t)(q0 + l16) * O3_;                  // Q row
  const float scale = 0.044194173824159216f;                             // 1/sqrt(512)

  v16bf qa[16];
#pragma unroll
  for (int i = 0; i < 16; ++i) qa[i] = load_a16(arow, i * 32, half);

  // Phase 1: streaming row-max.
  float mrow[8];
#pragma unroll
  for (int r = 0; r < 8; ++r) mrow[r] = -3.0e38f;
  for (int kt = 0; kt < 64; ++kt) {
    const __bf16* brow = base + (size_t)(kt * 16 + l16) * O3_ + C_;      // K row
    v8f acc = {};
#pragma unroll
    for (int i = 0; i < 16; ++i)
      acc = wmma_bf16(qa[i], load_b16(brow, i * 32, half), acc);
#pragma unroll
    for (int r = 0; r < 8; ++r) mrow[r] = fmaxf(mrow[r], acc[r] * scale);
  }
#pragma unroll
  for (int r = 0; r < 8; ++r)
#pragma unroll
    for (int off = 8; off > 0; off >>= 1)
      mrow[r] = fmaxf(mrow[r], __shfl_xor(mrow[r], off, 32));

  // Phase 2: recompute S, exp, accumulate rowsum, write P (row-contiguous via LDS).
  float rsum[8];
#pragma unroll
  for (int r = 0; r < 8; ++r) rsum[r] = 0.f;
  __shared__ __attribute__((aligned(32))) __bf16 lds[4][256];
  for (int kt = 0; kt < 64; ++kt) {
    const __bf16* brow = base + (size_t)(kt * 16 + l16) * O3_ + C_;
    v8f acc = {};
#pragma unroll
    for (int i = 0; i < 16; ++i)
      acc = wmma_bf16(qa[i], load_b16(brow, i * 32, half), acc);
#pragma unroll
    for (int r = 0; r < 8; ++r) {
      float p = __expf(acc[r] * scale - mrow[r]);
      rsum[r] += p;
      lds[wave][(r + 8 * half) * 16 + l16] = (__bf16)p;
    }
    __syncthreads();
    if (half == 0) {
      v16bf rowv = *(const v16bf*)&lds[wave][l16 * 16];
      *(v16bf*)(P + ((size_t)b * N_ + q0 + l16) * N_ + kt * 16) = rowv;
    }
    __syncthreads();
  }
#pragma unroll
  for (int r = 0; r < 8; ++r)
#pragma unroll
    for (int off = 8; off > 0; off >>= 1)
      rsum[r] += __shfl_xor(rsum[r], off, 32);
  if (l16 == 0) {
#pragma unroll
    for (int r = 0; r < 8; ++r) rowsum[b * N_ + q0 + 8 * half + r] = rsum[r];
  }
}

// ---------- Kernel 6: O = (P/rowsum) * V -> O[b][q][c] (bf16), 2x2 blocked ----------
__global__ void k_attn_out(const __bf16* __restrict__ P, const __bf16* __restrict__ VT,
                           const float* __restrict__ rowsum, __bf16* __restrict__ O) {
  const int wave = threadIdx.x >> 5;
  const int wid  = blockIdx.x * 4 + wave;
  const int lane = threadIdx.x & 31, half = lane >> 4, l16 = lane & 15;
  const int cp = wid & 15;                  // 512/32
  const int qp = (wid >> 4) & 31;           // 1024/32
  const int b  = wid >> 9;
  const int c0 = cp * 32, q0 = qp * 32;
  const __bf16* ar0 = P  + ((size_t)b * N_ + q0 + l16) * N_;     // A: M = query
  const __bf16* ar1 = ar0 + (size_t)16 * N_;
  const __bf16* br0 = VT + ((size_t)b * C_ + c0 + l16) * N_;     // B: N = channel
  const __bf16* br1 = br0 + (size_t)16 * N_;
  v8f c00 = {}, c01 = {}, c10 = {}, c11 = {};
#pragma unroll
  for (int k = 0; k < N_; k += 32) {
    v16bf a0 = load_a16(ar0, k, half);
    v16bf a1 = load_a16(ar1, k, half);
    v16bf b0 = load_b16(br0, k, half);
    v16bf b1 = load_b16(br1, k, half);
    c00 = wmma_bf16(a0, b0, c00);   // (q0..,   c0..)
    c01 = wmma_bf16(a0, b1, c01);   // (q0..,   c0+16..)
    c10 = wmma_bf16(a1, b0, c10);   // (q0+16.., c0..)
    c11 = wmma_bf16(a1, b1, c11);   // (q0+16.., c0+16..)
  }
  const float* rs0 = rowsum + b * N_ + q0 + half * 8;
  const float* rs1 = rs0 + 16;
  __shared__ __attribute__((aligned(64))) __bf16 lds[4][512];    // 16 rows x 32 ch per wave
  // rows q0..q0+15
#pragma unroll
  for (int r = 0; r < 8; ++r) {
    float inv = 1.0f / rs0[r];
    lds[wave][(r + 8 * half) * 32 + l16]      = (__bf16)(c00[r] * inv);
    lds[wave][(r + 8 * half) * 32 + 16 + l16] = (__bf16)(c01[r] * inv);
  }
  __syncthreads();
  if (half == 0) {
    const __bf16* lp = &lds[wave][l16 * 32];
    __bf16* dp = O + ((size_t)b * N_ + q0 + l16) * C_ + c0;
    *(v16bf*)(dp)      = *(const v16bf*)(lp);
    *(v16bf*)(dp + 16) = *(const v16bf*)(lp + 16);
  }
  __syncthreads();
  // rows q0+16..q0+31
#pragma unroll
  for (int r = 0; r < 8; ++r) {
    float inv = 1.0f / rs1[r];
    lds[wave][(r + 8 * half) * 32 + l16]      = (__bf16)(c10[r] * inv);
    lds[wave][(r + 8 * half) * 32 + 16 + l16] = (__bf16)(c11[r] * inv);
  }
  __syncthreads();
  if (half == 0) {
    const __bf16* lp = &lds[wave][l16 * 32];
    __bf16* dp = O + ((size_t)b * N_ + q0 + 16 + l16) * C_ + c0;
    *(v16bf*)(dp)      = *(const v16bf*)(lp);
    *(v16bf*)(dp + 16) = *(const v16bf*)(lp + 16);
  }
}

// ---------- Kernel 7: out = Wout*O + bias + x (fp32 [b,c,h,w]), 2x2 blocked ----------
__global__ void k_proj_out(const __bf16* __restrict__ O, const __bf16* __restrict__ wob,
                           const float* __restrict__ bout, const float* __restrict__ x,
                           float* __restrict__ out) {
  const int wid  = blockIdx.x * (blockDim.x >> 5) + (threadIdx.x >> 5);
  const int lane = threadIdx.x & 31, half = lane >> 4, l16 = lane & 15;
  const int op = wid & 15;                  // 512/32
  const int tp = (wid >> 4) & 31;           // 1024/32
  const int b  = wid >> 9;
  const int o0 = op * 32, t0 = tp * 32;
  const __bf16* ar0 = O   + ((size_t)b * N_ + t0 + l16) * C_;    // A: M = token
  const __bf16* ar1 = ar0 + (size_t)16 * C_;
  const __bf16* br0 = wob + (size_t)(o0 + l16) * C_;             // B: N = o
  const __bf16* br1 = br0 + (size_t)16 * C_;
  v8f c00 = {}, c01 = {}, c10 = {}, c11 = {};
#pragma unroll
  for (int k = 0; k < C_; k += 32) {
    v16bf a0 = load_a16(ar0, k, half);
    v16bf a1 = load_a16(ar1, k, half);
    v16bf b0 = load_b16(br0, k, half);
    v16bf b1 = load_b16(br1, k, half);
    c00 = wmma_bf16(a0, b0, c00);   // (t0..,    o0+l16)
    c01 = wmma_bf16(a0, b1, c01);   // (t0..,    o0+16+l16)
    c10 = wmma_bf16(a1, b0, c10);   // (t0+16.., o0+l16)
    c11 = wmma_bf16(a1, b1, c11);   // (t0+16.., o0+16+l16)
  }
  const v8f* accs[4] = { &c00, &c01, &c10, &c11 };
#pragma unroll
  for (int q = 0; q < 4; ++q) {
    const int o  = o0 + ((q & 1) ? 16 : 0) + l16;
    const int tb = t0 + ((q & 2) ? 16 : 0);
    const float bias = bout[o];
    const size_t off = ((size_t)b * C_ + o) * N_ + tb + half * 8;
    v4f x0 = *(const v4f*)(x + off);
    v4f x1 = *(const v4f*)(x + off + 4);
    const v8f& a = *accs[q];
    v4f lo, hi;
#pragma unroll
    for (int r = 0; r < 4; ++r) { lo[r] = a[r] + bias + x0[r]; hi[r] = a[r + 4] + bias + x1[r]; }
    *(v4f*)(out + off)     = lo;
    *(v4f*)(out + off + 4) = hi;
  }
}

extern "C" void kernel_launch(void* const* d_in, const int* in_sizes, int n_in,
                              void* d_out, int out_size, void* d_ws, size_t ws_size,
                              hipStream_t stream) {
  (void)in_sizes; (void)n_in; (void)out_size; (void)ws_size;
  const float* x      = (const float*)d_in[0];
  const float* gscale = (const float*)d_in[3];
  const float* gbias  = (const float*)d_in[4];
  const float* wqkv   = (const float*)d_in[5];
  const float* bqkv   = (const float*)d_in[6];
  const float* wout   = (const float*)d_in[7];
  const float* bout   = (const float*)d_in[8];
  float* out = (float*)d_out;

  char* ws = (char*)d_ws;
  size_t off = 0;
  auto alloc = [&](size_t bytes) -> void* {
    void* p = ws + off;
    off += (bytes + 255) & ~(size_t)255;
    return p;
  };
  float*  meanp  = (float*) alloc((size_t)B_ * G_ * 4);
  float*  rstdp  = (float*) alloc((size_t)B_ * G_ * 4);
  __bf16* wqb    = (__bf16*)alloc((size_t)O3_ * C_ * 2);
  __bf16* wob    = (__bf16*)alloc((size_t)C_ * C_ * 2);
  __bf16* xnT    = (__bf16*)alloc((size_t)B_ * N_ * C_ * 2);
  __bf16* qkvT   = (__bf16*)alloc((size_t)B_ * N_ * O3_ * 2);
  __bf16* VT     = (__bf16*)alloc((size_t)B_ * C_ * N_ * 2);
  __bf16* P      = (__bf16*)alloc((size_t)B_ * N_ * N_ * 2);
  float*  rowsum = (float*) alloc((size_t)B_ * N_ * 4);
  __bf16* Obuf   = (__bf16*)alloc((size_t)B_ * N_ * C_ * 2);

  k_cvt_weights   <<<512, 256, 0, stream>>>(wqkv, wout, wqb, wob);
  k_gn_stats      <<<B_ * G_, 256, 0, stream>>>(x, meanp, rstdp);
  k_norm_transpose<<<B_ * 16 * 32, 256, 0, stream>>>(x, meanp, rstdp, gscale, gbias, xnT);
  k_qkv_gemm      <<<(B_ * 32 * 48) / 8, 256, 0, stream>>>(wqb, xnT, bqkv, qkvT);
  k_vt            <<<B_ * 16 * 32, 256, 0, stream>>>(qkvT, VT);
  k_attn_softmax  <<<(B_ * 64) / 4, 128, 0, stream>>>(qkvT, P, rowsum);
  k_attn_out      <<<(B_ * 32 * 16) / 4, 128, 0, stream>>>(P, VT, rowsum, Obuf);
  k_proj_out      <<<(B_ * 32 * 16) / 8, 256, 0, stream>>>(Obuf, wob, bout, x, out);
}
